// AdditiveAttention_9337258902460
// MI455X (gfx1250) — matrix operand order repaired and checked by reading, not compiled, not run on hardware
//
#include <hip/hip_runtime.h>
#include <hip/hip_bf16.h>

// B=16, LQ=1, LK=8192, D=1024, H=1024
// scores[b,k] = sum_h tanh( (q@Wq)[b,h] + (keys[b,k,:]@Wk[:,h]) ) * wv[h]

#define BB   16
#define LK   8192
#define DD   1024
#define HH   1024
#define M_TILE 64
#define LDS_STRIDE 1032   // 1024 + 8 pad (elements) -> row stride 2064B ≡ 16 mod 256: conflict-free

typedef __attribute__((ext_vector_type(16))) __bf16          v16bf;
typedef __attribute__((ext_vector_type(8)))  float           v8f;
typedef __attribute__((ext_vector_type(8)))  unsigned short  u16x8;

__device__ __forceinline__ unsigned short f2bf(float f) {
    // round-to-nearest-even f32 -> bf16
    unsigned u = __float_as_uint(f);
    u += 0x7FFFu + ((u >> 16) & 1u);
    return (unsigned short)(u >> 16);
}

__device__ __forceinline__ float fast_tanh(float x) {
    float t;
    // gfx1250 transcendental op; ISA requires 1 independent op before result use
    asm volatile("v_tanh_f32 %0, %1\n\tv_nop" : "=v"(t) : "v"(x));
    return t;
}

__device__ __forceinline__ v16bf join16(u16x8 lo, u16x8 hi) {
    return __builtin_bit_cast(v16bf,
        __builtin_shufflevector(lo, hi, 0,1,2,3,4,5,6,7,8,9,10,11,12,13,14,15));
}

// ---------------------------------------------------------------- prep A:
// qp[b,h] = sum_d queries[b,d] * Wq[d,h]      (16x1024 outputs, trivial cost)
__global__ void qproj_kernel(const float* __restrict__ q,
                             const float* __restrict__ Wq,
                             float* __restrict__ qp) {
    int t = blockIdx.x * blockDim.x + threadIdx.x;   // 0..16383
    int b = t >> 10;
    int h = t & (HH - 1);
    const float* qrow = q + (size_t)b * DD;
    float s = 0.f;
    for (int d = 0; d < DD; ++d)
        s = fmaf(qrow[d], Wq[(size_t)d * HH + h], s);
    qp[t] = s;
}

// ---------------------------------------------------------------- prep B:
// Wt[h,d] = bf16(Wk[d,h])  -- transposed so each WMMA B-lane reads two runs of 8
// consecutive K values of its column (matches CDNA5 ISA B layout).
__global__ void wk_transpose_bf16(const float* __restrict__ Wk,
                                  unsigned short* __restrict__ Wt) {
    __shared__ float tile[32][33];
    const int tx = threadIdx.x & 31;       // 0..31
    const int ty = threadIdx.x >> 5;       // 0..7
    const int d0 = (blockIdx.x & 31) * 32;
    const int h0 = (blockIdx.x >> 5) * 32;
#pragma unroll
    for (int i = 0; i < 32; i += 8)
        tile[ty + i][tx] = Wk[(size_t)(d0 + ty + i) * HH + (h0 + tx)];
    __syncthreads();
#pragma unroll
    for (int i = 0; i < 32; i += 8)
        Wt[(size_t)(h0 + ty + i) * DD + (d0 + tx)] = f2bf(tile[tx][ty + i]);
}

// ---------------------------------------------------------------- main:
// One block owns 64 key rows x all 1024 H columns (fully fused, no global
// atomics). keys tile staged once in LDS as bf16, W_k streamed bf16 from L2.
__global__ void additive_attn_main(const float* __restrict__ keys,
                                   const unsigned short* __restrict__ Wt,  // bf16 [H][D]
                                   const float* __restrict__ qp,           // [B][H]
                                   const float* __restrict__ wv,           // [H]
                                   float* __restrict__ out) {              // [B][LK]
    extern __shared__ unsigned short lds[];                 // [64][1032] bf16 keys
    float* sScores = (float*)(lds + M_TILE * LDS_STRIDE);   // [64] f32

    const int tid   = threadIdx.x;
    const int wave  = tid >> 5;
    const int lane  = tid & 31;
    const int nlane = lane & 15;     // C/D column within tile, B column
    const int sel   = lane >> 4;     // half-wave select for A/B K-split
    const int row0  = blockIdx.x * M_TILE;       // global key row
    const int b     = row0 / LK;                 // 64 | 8192, never straddles b
    const float* keysBase = keys + (size_t)row0 * DD;

    if (tid < M_TILE) sScores[tid] = 0.f;

    // ---- stage keys tile: 64x1024 f32 -> bf16 LDS (each element read once)
#pragma unroll 4
    for (int it = 0; it < (M_TILE * DD) / (256 * 4); ++it) {
        int e = (it * 256 + tid) * 4;
        int r = e >> 10, c = e & (DD - 1);
        const float4 v = *reinterpret_cast<const float4*>(keysBase + (size_t)r * DD + c);
        uint2 pk;
        pk.x = (unsigned)f2bf(v.x) | ((unsigned)f2bf(v.y) << 16);
        pk.y = (unsigned)f2bf(v.z) | ((unsigned)f2bf(v.w) << 16);
        *reinterpret_cast<uint2*>(&lds[r * LDS_STRIDE + c]) = pk;   // one ds_store_b64
    }
    __syncthreads();

    // A-fragment LDS bases: depend only on (m, nlane, sel) -> hoisted so every
    // ds_load in the K-loop uses an immediate offset (k0*2 <= 2016 bytes).
    const unsigned short* aBase[4];
#pragma unroll
    for (int m = 0; m < 4; ++m)
        aBase[m] = &lds[(m * 16 + nlane) * LDS_STRIDE + sel * 8];

    float rowsum[4][8];
#pragma unroll
    for (int m = 0; m < 4; ++m)
#pragma unroll
        for (int r = 0; r < 8; ++r) rowsum[m][r] = 0.f;

    // ---- 8 column iterations: wave w covers h = nIter*128 + w*16 + [0,16)
#pragma unroll 1
    for (int nIter = 0; nIter < 8; ++nIter) {
        const int h0  = nIter * 128 + wave * 16;
        const int col = h0 + nlane;
        // B column = Wt row: per lane K k0+8*sel..+7 and k0+16+8*sel..+7
        // (matches A table order), 2x16B global reads.
        const u16x8* brow = reinterpret_cast<const u16x8*>(Wt + (size_t)col * DD + sel * 8);

        v8f acc[4] = {};   // 4 M-subtiles x 16x16 f32 accum

        // software-pipelined B: fetch k0+32 while WMMAs for k0 run
        v16bf bcur = join16(brow[0], brow[2]);

#pragma unroll 4
        for (int k0 = 0; k0 < DD; k0 += 32) {
            // next B fragment (last step over-reads 32B into qp ws region: benign)
            const u16x8* bnp = brow + (k0 + 32) / 8;
            u16x8 bnlo = bnp[0];
            u16x8 bnhi = bnp[2];
            __builtin_prefetch(brow + (k0 + 256) / 8, 0, 1);   // global_prefetch_b8

            // all 4 A fragments into DISTINCT regs, then 4 back-to-back WMMAs
            // (no overlapping VGPRs -> no WMMA->VALU hazard NOPs, single wait)
            v16bf a0, a1, a2, a3;
            {
                const unsigned short* p0 = aBase[0] + k0;
                const unsigned short* p1 = aBase[1] + k0;
                const unsigned short* p2 = aBase[2] + k0;
                const unsigned short* p3 = aBase[3] + k0;
                a0 = join16(*(const u16x8*)p0, *(const u16x8*)(p0 + 16));
                a1 = join16(*(const u16x8*)p1, *(const u16x8*)(p1 + 16));
                a2 = join16(*(const u16x8*)p2, *(const u16x8*)(p2 + 16));
                a3 = join16(*(const u16x8*)p3, *(const u16x8*)(p3 + 16));
            }
            acc[0] = __builtin_amdgcn_wmma_f32_16x16x32_bf16(
                false, a0, false, bcur, (short)0, acc[0], false, false);
            acc[1] = __builtin_amdgcn_wmma_f32_16x16x32_bf16(
                false, a1, false, bcur, (short)0, acc[1], false, false);
            acc[2] = __builtin_amdgcn_wmma_f32_16x16x32_bf16(
                false, a2, false, bcur, (short)0, acc[2], false, false);
            acc[3] = __builtin_amdgcn_wmma_f32_16x16x32_bf16(
                false, a3, false, bcur, (short)0, acc[3], false, false);

            bcur = join16(bnlo, bnhi);
        }

        // ---- fused epilogue: tanh(qp + acc) * wv, reduce over this column
        const float qv  = qp[b * HH + col];
        const float wvv = wv[col];
#pragma unroll
        for (int m = 0; m < 4; ++m)
#pragma unroll
            for (int r = 0; r < 8; ++r)
                rowsum[m][r] += fast_tanh(qv + acc[m][r]) * wvv;
    }

    // ---- reduce over the 16 columns of each half-wave (C layout:
    // vgpr r, lanes0-15 -> M=r ; lanes16-31 -> M=8+r), then across waves.
#pragma unroll
    for (int m = 0; m < 4; ++m)
#pragma unroll
        for (int r = 0; r < 8; ++r) {
            float v = rowsum[m][r];
            v += __shfl_xor(v, 1, 32);
            v += __shfl_xor(v, 2, 32);
            v += __shfl_xor(v, 4, 32);
            v += __shfl_xor(v, 8, 32);
            if (nlane == 0)
                atomicAdd(&sScores[m * 16 + r + 8 * sel], v);  // ds_add_f32
        }
    __syncthreads();
    if (tid < M_TILE) out[row0 + tid] = sScores[tid];
}

// ----------------------------------------------------------------
extern "C" void kernel_launch(void* const* d_in, const int* in_sizes, int n_in,
                              void* d_out, int out_size, void* d_ws, size_t ws_size,
                              hipStream_t stream) {
    const float* queries = (const float*)d_in[0];  // [16,1,1024]
    const float* keys    = (const float*)d_in[1];  // [16,8192,1024]
    const float* W_q     = (const float*)d_in[2];  // [1024,1024]
    const float* W_k     = (const float*)d_in[3];  // [1024,1024]
    const float* w_v     = (const float*)d_in[4];  // [1024,1]
    float* out = (float*)d_out;                    // [16,8192]

    unsigned short* Wt = (unsigned short*)d_ws;                       // 2 MB bf16
    float* qp = (float*)((char*)d_ws + (size_t)HH * DD * sizeof(unsigned short)); // 64 KB

    // prep: q projection + W_k transpose/convert (independent, tiny)
    qproj_kernel<<<(BB * HH) / 256, 256, 0, stream>>>(queries, W_q, qp);
    wk_transpose_bf16<<<(DD / 32) * (HH / 32), 256, 0, stream>>>(W_k, Wt);

    // main fused kernel: 2048 blocks x 256 threads, 132,352 B dynamic LDS
    const size_t ldsBytes = (size_t)M_TILE * LDS_STRIDE * sizeof(unsigned short)
                          + M_TILE * sizeof(float);
    additive_attn_main<<<(BB * LK) / M_TILE, 256, ldsBytes, stream>>>(
        keys, Wt, qp, w_v, out);
    (void)in_sizes; (void)n_in; (void)out_size; (void)ws_size;
}
